// SummaryAttention_38706245271874
// MI455X (gfx1250) — matrix operand order repaired and checked by reading, not compiled, not run on hardware
//
#include <hip/hip_runtime.h>
#include <hip/hip_bf16.h>

// Problem constants from the reference
#define B_  8
#define T_  64
#define S_  256
#define D_  512

typedef __attribute__((ext_vector_type(16))) __bf16 v16bf;
typedef __attribute__((ext_vector_type(8)))  __bf16 v8bf;
typedef __attribute__((ext_vector_type(8)))  float  v8f;
typedef __attribute__((ext_vector_type(2)))  int    v2i;

#if __has_builtin(__builtin_amdgcn_global_load_async_to_lds_b64) && \
    __has_builtin(__builtin_amdgcn_s_wait_asynccnt)
#define ASYNC_LDS 1
#else
#define ASYNC_LDS 0
#endif

// Async builtin operand types (discovered via compile probes):
//   arg0: v2i addrspace(1)*  (global source)
//   arg1: v2i addrspace(3)*  (LDS destination)
#define ASG64(p) ((__attribute__((address_space(1))) v2i*)(void*)(p))
#define ASL64(p) ((__attribute__((address_space(3))) v2i*)(p))

__device__ inline unsigned short f32_to_bf16_rne(float f) {
  unsigned u = __builtin_bit_cast(unsigned, f);
  unsigned r = u + 0x7FFFu + ((u >> 16) & 1u);
  return (unsigned short)(r >> 16);
}

__device__ inline float fast_tanh(float x) {
  // tanh(x) = 1 - 2/(exp(2x)+1); saturates correctly at +/-inf
  return 1.0f - 2.0f / (__expf(2.0f * x) + 1.0f);
}

// --- Stage 1a: running mean (cumsum/count) * mask -> bf16 acc [B*T, D] ---
__global__ __launch_bounds__(256) void prep_acc_kernel(
    const float* __restrict__ src, const float* __restrict__ masks,
    unsigned short* __restrict__ accb) {
  int idx = blockIdx.x * blockDim.x + threadIdx.x;  // b*D + d, 4096 total
  int b = idx >> 9;
  int d = idx & (D_ - 1);
  float sum = 0.f;
  for (int t = 0; t < T_; ++t) {
    sum += src[(b * T_ + t) * D_ + d];
    float a = (sum / (float)(t + 1)) * masks[b * T_ + t];
    accb[(b * T_ + t) * D_ + d] = f32_to_bf16_rne(a);
  }
}

// --- Stage 1b: Wq -> bf16, TRANSPOSED: wqt[n*D + k] = bf16(Wq[k*D + n]) ---
// Transposed storage makes every WMMA B-fragment two contiguous b128 loads.
__global__ __launch_bounds__(256) void conv_wq_kernel(
    const float* __restrict__ wq, unsigned short* __restrict__ wqt) {
  int i = blockIdx.x * blockDim.x + threadIdx.x;  // n*D + k (coalesced write)
  int n = i >> 9;
  int k = i & (D_ - 1);
  wqt[i] = f32_to_bf16_rne(wq[k * D_ + n]);
}

// --- Stage 1c: wsum[k] = sum_n Wv[k,n] ---
__global__ __launch_bounds__(256) void wsum_kernel(
    const float* __restrict__ wv, float* __restrict__ wsum) {
  int k = blockIdx.x * blockDim.x + threadIdx.x;  // 512 threads
  if (k < D_) {
    float s = 0.f;
    for (int n = 0; n < D_; ++n) s += wv[k * D_ + n];
    wsum[k] = s;
  }
}

// --- Stage 1d: per (b,s): p1 = mb.Wc[:D], p2 = mb.Wc[D:]  (wave per row) ---
__global__ __launch_bounds__(256) void pp_kernel(
    const float* __restrict__ mb, const float* __restrict__ wc,
    float* __restrict__ pp) {
  int gw = (blockIdx.x * blockDim.x + threadIdx.x) >> 5;  // 0..B*S-1
  int lane = threadIdx.x & 31;
  const float* row = mb + gw * D_;
  float a0 = 0.f, a1 = 0.f, a2 = 0.f, a3 = 0.f;
  for (int d = lane; d < D_; d += 32) {
    float m = row[d];
    a0 += m * wc[d * 2 + 0];
    a1 += m * wc[d * 2 + 1];
    a2 += m * wc[(D_ + d) * 2 + 0];
    a3 += m * wc[(D_ + d) * 2 + 1];
  }
#pragma unroll
  for (int off = 16; off > 0; off >>= 1) {
    a0 += __shfl_xor(a0, off, 32);
    a1 += __shfl_xor(a1, off, 32);
    a2 += __shfl_xor(a2, off, 32);
    a3 += __shfl_xor(a3, off, 32);
  }
  if (lane == 0) {
    pp[gw * 4 + 0] = a0;
    pp[gw * 4 + 1] = a1;
    pp[gw * 4 + 2] = a2;
    pp[gw * 4 + 3] = a3;
  }
}

__device__ inline v16bf load_frag16(const unsigned short* p0,
                                    const unsigned short* p1) {
  v8bf lo = *(const v8bf*)p0;   // 16B aligned contiguous
  v8bf hi = *(const v8bf*)p1;
  return __builtin_shufflevector(lo, hi, 0, 1, 2, 3, 4, 5, 6, 7,
                                 8, 9, 10, 11, 12, 13, 14, 15);
}

// --- Stage 2: ht = acc @ Wq + bq  via v_wmma_f32_16x16x32_bf16 ---
// M = B*T = 512, N = D = 512, K = D = 512. One 16x16 tile per wave.
__global__ __launch_bounds__(256) void wmma_ht_kernel(
    const unsigned short* __restrict__ accb, const unsigned short* __restrict__ wqt,
    const float* __restrict__ bq, float* __restrict__ ht) {
  const int lane  = threadIdx.x & 31;
  const int wave  = threadIdx.x >> 5;
  const int ntile = blockIdx.x;                 // 0..31
  const int mtile = blockIdx.y * 8 + wave;      // 0..31
  const int half  = lane >> 4;
  const int lm    = lane & 15;

  const int arow = mtile * 16 + lm;   // A-matrix row owned by this lane
  const int bcol = ntile * 16 + lm;   // B-matrix column owned by this lane

  const unsigned short* abase = accb + arow * D_ + half * 16;  // contiguous 16
  const unsigned short* bbase = wqt + bcol * D_ + half * 16;   // contiguous 16

  v8f c = {};
  for (int kb = 0; kb < D_; kb += 32) {
    // A fragment (16x32 bf16): lanes 0-15 -> K {kb..+15},
    //                          lanes 16-31 -> K {kb+16..+31}
    v16bf a = load_frag16(abase + kb, abase + kb + 8);
    // B fragment (32x16 bf16): element i -> K = kb + 16*half + i, N = bcol.
    // WqT row-major in K => 16 contiguous bf16 => two b128 loads.
    v16bf b = load_frag16(bbase + kb, bbase + kb + 8);

    c = __builtin_amdgcn_wmma_f32_16x16x32_bf16(
        /*neg_a=*/false, a, /*neg_b=*/false, b,
        /*c_mod=*/(short)0, c, /*reuse_a=*/false, /*reuse_b=*/false);
  }

  // C/D layout: VGPR r -> M = r + 8*half, N = lane%16
#pragma unroll
  for (int r = 0; r < 8; ++r) {
    int m = mtile * 16 + r + 8 * half;
    ht[m * D_ + bcol] = c[r] + bq[bcol];
  }
}

// --- Stage 3: gate + fuse. Block per (b,s); wave per t; lanes stride D. ---
__global__ __launch_bounds__(256) void gate_fuse_kernel(
    const float* __restrict__ ht, const float* __restrict__ mb,
    const float* __restrict__ wsum, const float* __restrict__ pp,
    const float* __restrict__ bc, float* __restrict__ out) {
  __shared__ float mbld[D_];
  __shared__ float wsl[D_];
  const int bs = blockIdx.x;      // 0..B*S-1
  const int b  = bs >> 8;
  const int s  = bs & (S_ - 1);
  const int tid = threadIdx.x;

  const float* mrow = mb + (size_t)bs * D_;
#if ASYNC_LDS
  // gfx1250 async copy: global -> LDS without touching VGPR data path.
  __builtin_amdgcn_global_load_async_to_lds_b64(
      ASG64(mrow + 2 * tid), ASL64(&mbld[2 * tid]), 0, 0);
  __builtin_amdgcn_global_load_async_to_lds_b64(
      ASG64(wsum + 2 * tid), ASL64(&wsl[2 * tid]), 0, 0);
  __builtin_amdgcn_s_wait_asynccnt(0);
#else
  mbld[tid]       = mrow[tid];
  mbld[tid + 256] = mrow[tid + 256];
  wsl[tid]        = wsum[tid];
  wsl[tid + 256]  = wsum[tid + 256];
#endif
  __syncthreads();

  const int lane = tid & 31;
  const int wave = tid >> 5;
  const float p10 = pp[bs * 4 + 0], p11 = pp[bs * 4 + 1];
  const float p20 = pp[bs * 4 + 2], p21 = pp[bs * 4 + 3];
  const float bc0 = bc[0], bc1 = bc[1];

  for (int it = 0; it < T_ / 8; ++it) {
    const int t = it * 8 + wave;
    const float* hrow = ht + (size_t)(b * T_ + t) * D_;
    if (it < T_ / 8 - 1)
      __builtin_prefetch(hrow + 8 * D_ + lane, 0, 1);  // next t for this wave
    float partial = 0.f;
#pragma unroll
    for (int j = 0; j < D_ / 32; ++j) {
      int d = j * 32 + lane;  // coalesced across the wave
      partial += fast_tanh(hrow[d] + mbld[d]) * wsl[d];
    }
#pragma unroll
    for (int off = 16; off > 0; off >>= 1)
      partial += __shfl_xor(partial, off, 32);
    if (lane == 0) {
      float g = partial;
      float* o = out + (size_t)((b * T_ + t) * S_ + s) * 2;
      o[0] = p10 + g * p20 + bc0;
      o[1] = p11 + g * p21 + bc1;
    }
  }
}

extern "C" void kernel_launch(void* const* d_in, const int* in_sizes, int n_in,
                              void* d_out, int out_size, void* d_ws, size_t ws_size,
                              hipStream_t stream) {
  (void)in_sizes; (void)n_in; (void)out_size; (void)ws_size;

  const float* source  = (const float*)d_in[0];  // [B,T,D]
  const float* membank = (const float*)d_in[1];  // [B,S,D]
  const float* masks   = (const float*)d_in[2];  // [B,T]
  const float* Wq      = (const float*)d_in[3];  // [D,D]
  const float* bq      = (const float*)d_in[4];  // [D]
  const float* Wv      = (const float*)d_in[5];  // [D,D]
  const float* Wc      = (const float*)d_in[6];  // [2D,2]
  const float* bc      = (const float*)d_in[7];  // [2]
  float* out = (float*)d_out;                    // [B,T,S,2]

  // Workspace carve-up (~2.03 MB total)
  char* ws = (char*)d_ws;
  unsigned short* accb = (unsigned short*)ws;                    // 512 KB
  unsigned short* wqt  = (unsigned short*)(ws + 524288);         // 512 KB (Wq^T bf16)
  float* ht   = (float*)(ws + 1048576);                          // 1 MB
  float* wsum = (float*)(ws + 2097152);                          // 2 KB
  float* pp   = (float*)(ws + 2099200);                          // 32 KB

  // Stage 1: prep
  prep_acc_kernel<<<dim3(16), dim3(256), 0, stream>>>(source, masks, accb);
  conv_wq_kernel<<<dim3(1024), dim3(256), 0, stream>>>(Wq, wqt);
  wsum_kernel<<<dim3(2), dim3(256), 0, stream>>>(Wv, wsum);
  pp_kernel<<<dim3(256), dim3(256), 0, stream>>>(membank, Wc, pp);

  // Stage 2: ht = acc @ Wq + bq (WMMA bf16, f32 accumulate)
  wmma_ht_kernel<<<dim3(32, 4), dim3(256), 0, stream>>>(accb, wqt, bq, ht);

  // Stage 3: gate (tanh dot with Wv row-sums) + fused output
  gate_fuse_kernel<<<dim3(B_ * S_), dim3(256), 0, stream>>>(
      ht, membank, wsum, pp, bc, out);
}
